// FinchB2TimeMix_25391846654568
// MI455X (gfx1250) — matrix operand order repaired and checked by reading, not compiled, not run on hardware
//
#include <hip/hip_runtime.h>
#include <hip/hip_bf16.h>

// ---- problem constants (fixed by reference) ----
#define TT 4096
#define BB 2
#define CC 2048
#define BT (BB * TT)
#define HH 64
#define NHEADS (CC / HH)   // 32

typedef __attribute__((ext_vector_type(16))) __bf16 v16bf;
typedef __attribute__((ext_vector_type(8)))  float  v8f;
typedef __attribute__((ext_vector_type(4)))  int    v4i;

union Frag16 {
    unsigned int u32[8];
    v16bf v;
};

__device__ __forceinline__ unsigned short f2bf(float f) {
    unsigned int u = __float_as_uint(f);
    unsigned int r = (u + 0x7FFFu + ((u >> 16) & 1u)) >> 16;  // RNE
    return (unsigned short)r;
}

// ---------------------------------------------------------------------
// CDNA5 async global->LDS staging (ASYNCcnt path), builtin or asm
// Builtin signature (from hipcc diagnostic): pointers to int4, global
// source first, LDS destination second, then imm offset / imm cpol.
// ---------------------------------------------------------------------
__device__ __forceinline__ void async_b128(const unsigned short* g, unsigned short* l) {
#if __has_builtin(__builtin_amdgcn_global_load_async_to_lds_b128)
    __builtin_amdgcn_global_load_async_to_lds_b128(
        (__attribute__((address_space(1))) v4i*)(unsigned long long)(uintptr_t)g,
        (__attribute__((address_space(3))) v4i*)(void*)l,
        0, 0);
#else
    unsigned loff =
        (unsigned)(uintptr_t)(__attribute__((address_space(3))) char*)(void*)l;
    unsigned long long ga = (unsigned long long)(uintptr_t)g;
    asm volatile("global_load_async_to_lds_b128 %0, %1, off"
                 :: "v"(loff), "v"(ga) : "memory");
#endif
}

__device__ __forceinline__ void wait_async0() {
#if __has_builtin(__builtin_amdgcn_s_wait_asynccnt)
    __builtin_amdgcn_s_wait_asynccnt(0);
#else
    asm volatile("s_wait_asynccnt 0" ::: "memory");
#endif
}

// =====================================================================
// Weight conversion kernels
// =====================================================================
__global__ void cvt_bf16_kernel(const float* __restrict__ src,
                                unsigned short* __restrict__ dst, size_t n) {
    size_t i = (size_t)blockIdx.x * 256 + threadIdx.x;
    if (i < n) dst[i] = f2bf(src[i]);
}

// src [R][Cc] -> dst [Cc][R]   (so the GEMM always sees W as [n][k])
__global__ void cvt_bf16_t_kernel(const float* __restrict__ src,
                                  unsigned short* __restrict__ dst, int R, int Cc) {
    size_t n = (size_t)R * Cc;
    size_t i = (size_t)blockIdx.x * 256 + threadIdx.x;
    if (i < n) {
        int r = (int)(i / Cc);
        int c = (int)(i % Cc);
        dst[(size_t)c * R + r] = f2bf(src[i]);
    }
}

// =====================================================================
// Token shift + LoRA-in:  xxx[g,160] = tanh((x + dxprev*maa_x) @ maa_w1)
// =====================================================================
__global__ __launch_bounds__(256)
void lora_in_kernel(const float* __restrict__ x, const float* __restrict__ shift,
                    const float* __restrict__ maa_x, const float* __restrict__ maa_w1,
                    float* __restrict__ xxx) {
    __shared__ float xm[CC];
    int g = blockIdx.x;
    int b = g / TT, t = g % TT;
#pragma unroll
    for (int i = 0; i < CC / 256; ++i) {
        int c = threadIdx.x + i * 256;
        float xc = x[(size_t)g * CC + c];
        float prev = (t == 0) ? shift[b * CC + c] : x[(size_t)(g - 1) * CC + c];
        xm[c] = xc + (prev - xc) * maa_x[c];
    }
    __syncthreads();
    int f = threadIdx.x;
    if (f < 160) {
        float acc = 0.f;
#pragma unroll 8
        for (int c = 0; c < CC; ++c) acc += xm[c] * maa_w1[c * 160 + f];
        xxx[(size_t)g * 160 + f] = tanhf(acc);
    }
}

// =====================================================================
// Mix-out: build xr,xk,xv,xw,xv2 in bf16 (GEMM A operands)
// =====================================================================
__global__ __launch_bounds__(256)
void mix_out_kernel(const float* __restrict__ x, const float* __restrict__ shift,
                    const float* __restrict__ xxx, const float* __restrict__ maa_w2,
                    const float* __restrict__ maa_r, const float* __restrict__ maa_k,
                    const float* __restrict__ maa_v, const float* __restrict__ maa_w,
                    const float* __restrict__ maa_v2,
                    unsigned short* __restrict__ ar, unsigned short* __restrict__ ak,
                    unsigned short* __restrict__ av, unsigned short* __restrict__ aw,
                    unsigned short* __restrict__ av2) {
    __shared__ float xl[160];
    int g = blockIdx.x;
    int b = g / TT, t = g % TT;
    if (threadIdx.x < 160) xl[threadIdx.x] = xxx[(size_t)g * 160 + threadIdx.x];
    __syncthreads();
#pragma unroll
    for (int i = 0; i < CC / 256; ++i) {
        int c = threadIdx.x + i * 256;
        size_t idx = (size_t)g * CC + c;
        float xc = x[idx];
        float prev = (t == 0) ? shift[b * CC + c] : x[idx - CC];
        float dx = prev - xc;
        float mv[5];
#pragma unroll
        for (int w = 0; w < 5; ++w) {
            float a = 0.f;
#pragma unroll
            for (int l = 0; l < 32; ++l)
                a += xl[w * 32 + l] * maa_w2[(size_t)(w * 32 + l) * CC + c];
            mv[w] = a;
        }
        ar [idx] = f2bf(xc + dx * (maa_r [c] + mv[0]));
        ak [idx] = f2bf(xc + dx * (maa_k [c] + mv[1]));
        av [idx] = f2bf(xc + dx * (maa_v [c] + mv[2]));
        aw [idx] = f2bf(xc + dx * (maa_w [c] + mv[3]));
        av2[idx] = f2bf(xc + dx * (maa_v2[c] + mv[4]));
    }
}

// =====================================================================
// WMMA GEMM:  Out[m,n] = sum_k A[m,k] * W[n,k]
//   A : bf16 [M=8192][K] row-major      W : bf16 [N][K] row-major
//   128x128 block tile, 8 waves (4M x 2N), each wave 32x64 via 2x4 WMMAs
//   Tiles staged with GLOBAL_LOAD_ASYNC_TO_LDS_B128, double buffered.
//   NOTE: B rows beyond N are loaded unguarded; every B operand passed in
//   has >= 128 rows of in-workspace memory behind it, and stores are
//   guarded by nn < N, so garbage rows never reach the output.
//   epi: 0 = store f32, 1 = tanh -> bf16, 2 = accumulate into f32
// =====================================================================
#define TILE_M 128
#define TILE_N 128
#define TILE_K 32
#define LDSS   40   // padded LDS row stride (elements) — bank-conflict-free

__global__ __launch_bounds__(256)
void gemm_bf16_wmma(const unsigned short* __restrict__ A,
                    const unsigned short* __restrict__ Bw,
                    int N, int K,
                    float* __restrict__ Cf,
                    unsigned short* __restrict__ Cb,
                    int epi) {
    __shared__ __align__(16) unsigned short As[2][TILE_M * LDSS];
    __shared__ __align__(16) unsigned short Bs[2][TILE_N * LDSS];

    const int tid  = threadIdx.x;
    const int lane = tid & 31;
    const int wid  = tid >> 5;
    const int m_wave = (wid & 3) * 32;
    const int n_wave = (wid >> 2) * 64;
    const int m0 = blockIdx.y * TILE_M;
    const int n0 = blockIdx.x * TILE_N;

    // per-thread staging coords: two 16B chunks each for A and B
    const int ar0 = tid >> 2,         ac0 = (tid & 3) * 8;
    const int ar1 = (tid + 256) >> 2, ac1 = ((tid + 256) & 3) * 8;

    v8f acc[2][4];
#pragma unroll
    for (int i = 0; i < 2; ++i)
#pragma unroll
        for (int j = 0; j < 4; ++j)
#pragma unroll
            for (int e = 0; e < 8; ++e) acc[i][j][e] = 0.0f;

    const int NK = K / TILE_K;

    // ---- stage tile 0 ----
    {
        async_b128(A  + (size_t)(m0 + ar0) * K + ac0, &As[0][ar0 * LDSS + ac0]);
        async_b128(A  + (size_t)(m0 + ar1) * K + ac1, &As[0][ar1 * LDSS + ac1]);
        async_b128(Bw + (size_t)(n0 + ar0) * K + ac0, &Bs[0][ar0 * LDSS + ac0]);
        async_b128(Bw + (size_t)(n0 + ar1) * K + ac1, &Bs[0][ar1 * LDSS + ac1]);
    }
    wait_async0();
    __syncthreads();

    int cur = 0;
    for (int kt = 0; kt < NK; ++kt) {
        // ---- prefetch next tile into the other buffer ----
        if (kt + 1 < NK) {
            int k0 = (kt + 1) * TILE_K;
            int nb = cur ^ 1;
            async_b128(A  + (size_t)(m0 + ar0) * K + k0 + ac0, &As[nb][ar0 * LDSS + ac0]);
            async_b128(A  + (size_t)(m0 + ar1) * K + k0 + ac1, &As[nb][ar1 * LDSS + ac1]);
            async_b128(Bw + (size_t)(n0 + ar0) * K + k0 + ac0, &Bs[nb][ar0 * LDSS + ac0]);
            async_b128(Bw + (size_t)(n0 + ar1) * K + k0 + ac1, &Bs[nb][ar1 * LDSS + ac1]);
        }

        // ---- A frags: ISA 16-bit A 16x32 layout ----
        const unsigned short* Ac = As[cur];
        const unsigned short* Bc = Bs[cur];
        Frag16 afr[2];
#pragma unroll
        for (int i = 0; i < 2; ++i) {
            int m   = m_wave + i * 16 + (lane & 15);
            int upo = ((lane >> 4) & 1) << 3;   // 0 or 8
#pragma unroll
            for (int q = 0; q < 8; ++q) {
                int kb = ((q < 4) ? 0 : 16) + upo + ((q & 3) << 1);
                afr[i].u32[q] = *(const unsigned int*)(Ac + m * LDSS + kb);
            }
        }
        // ---- B frags: ISA 16-bit B 32x16 layout ----
        Frag16 bfr[4];
#pragma unroll
        for (int j = 0; j < 4; ++j) {
            int n   = n_wave + j * 16 + (lane & 15);
            int kup = ((lane >> 4) & 1) << 3;   // 0 or 8
#pragma unroll
            for (int q = 0; q < 8; ++q) {
                int kb = ((q < 4) ? 0 : 16) + kup + ((q & 3) << 1);
                bfr[j].u32[q] = *(const unsigned int*)(Bc + n * LDSS + kb);
            }
        }
#pragma unroll
        for (int i = 0; i < 2; ++i)
#pragma unroll
            for (int j = 0; j < 4; ++j)
                acc[i][j] = __builtin_amdgcn_wmma_f32_16x16x32_bf16(
                    false, afr[i].v, false, bfr[j].v,
                    (short)0, acc[i][j], false, false);

        if (kt + 1 < NK) {
            wait_async0();      // my async writes for next tile are in LDS
            __syncthreads();    // everyone's are; everyone done reading cur
        }
        cur ^= 1;
    }

    // epilogue: D layout — VGPR e holds M = e (+8 for lanes 16-31), N = lane&15
    const int nloc = lane & 15;
    const int mup  = ((lane >> 4) & 1) * 8;
#pragma unroll
    for (int i = 0; i < 2; ++i) {
#pragma unroll
        for (int j = 0; j < 4; ++j) {
            int nn = n0 + n_wave + j * 16 + nloc;
            if (nn >= N) continue;
#pragma unroll
            for (int e = 0; e < 8; ++e) {
                int mm = m0 + m_wave + i * 16 + mup + e;
                float val = acc[i][j][e];
                size_t off = (size_t)mm * N + nn;
                if (epi == 0)      Cf[off] = val;
                else if (epi == 1) Cb[off] = f2bf(tanhf(val));
                else               Cf[off] += val;
            }
        }
    }
}

// =====================================================================
// decay = exp(-exp(time_decay + w_lora));  k *= (1 - decay)
// =====================================================================
__global__ __launch_bounds__(256)
void postmix_kernel(const float* __restrict__ td,
                    float* __restrict__ wdec, float* __restrict__ kbuf) {
    size_t idx = (size_t)blockIdx.x * 256 + threadIdx.x;
    int c = (int)(idx % CC);
    float w = td[c] + wdec[idx];
    float d = expf(-expf(w));
    wdec[idx] = d;
    kbuf[idx] *= (1.0f - d);
}

// =====================================================================
// WKV serial recurrence: one block per (b, head); lane j owns state col j
// =====================================================================
__global__ __launch_bounds__(64)
void wkv_kernel(const float* __restrict__ r, const float* __restrict__ k,
                const float* __restrict__ v, const float* __restrict__ v2,
                const float* __restrict__ dec, const float* __restrict__ st_in,
                float* __restrict__ y, float* __restrict__ st_out) {
    int head = blockIdx.x;              // b*NHEADS + n
    int b = head / NHEADS, n = head % NHEADS;
    int j = threadIdx.x;
    size_t sbase = (size_t)head * HH * HH;

    float s[HH];
#pragma unroll
    for (int h = 0; h < HH; ++h) s[h] = st_in[sbase + h * HH + j];

    __shared__ float rs[HH], ks[HH], ds[HH];
    for (int t = 0; t < TT; ++t) {
        size_t base = (size_t)(b * TT + t) * CC + n * HH;
        rs[j] = r[base + j];
        ks[j] = k[base + j];
        ds[j] = dec[base + j];
        __syncthreads();
        float vj = v[base + j];
        float acc = v2[base + j];
#pragma unroll
        for (int h = 0; h < HH; ++h) {
            acc  += rs[h] * s[h];
            s[h]  = s[h] * ds[h] + ks[h] * vj;
        }
        y[base + j] = acc;
        __syncthreads();
    }
#pragma unroll
    for (int h = 0; h < HH; ++h) st_out[sbase + h * HH + j] = s[h];
}

// =====================================================================
// LayerNorm -> bf16 (A operand for the Wo GEMM)
// =====================================================================
__global__ __launch_bounds__(256)
void ln_kernel(const float* __restrict__ y, const float* __restrict__ g,
               const float* __restrict__ bta, unsigned short* __restrict__ out) {
    __shared__ float red[256], red2[256];
    int t = blockIdx.x;
    float s = 0.f, s2 = 0.f;
#pragma unroll
    for (int i = 0; i < CC / 256; ++i) {
        float v = y[(size_t)t * CC + threadIdx.x + i * 256];
        s += v; s2 += v * v;
    }
    red[threadIdx.x] = s; red2[threadIdx.x] = s2;
    __syncthreads();
    for (int off = 128; off > 0; off >>= 1) {
        if (threadIdx.x < off) {
            red[threadIdx.x]  += red[threadIdx.x + off];
            red2[threadIdx.x] += red2[threadIdx.x + off];
        }
        __syncthreads();
    }
    float mu   = red[0] / CC;
    float var  = red2[0] / CC - mu * mu;
    float rstd = rsqrtf(var + 1e-5f);
#pragma unroll
    for (int i = 0; i < CC / 256; ++i) {
        int c = threadIdx.x + i * 256;
        float v = y[(size_t)t * CC + c];
        out[(size_t)t * CC + c] = f2bf((v - mu) * rstd * g[c] + bta[c]);
    }
}

// =====================================================================
// x[:, -1] -> tail of d_out
// =====================================================================
__global__ void tail_kernel(const float* __restrict__ x, float* __restrict__ out) {
    int i = blockIdx.x * 256 + threadIdx.x;
    if (i < BB * CC) {
        int b = i / CC, c = i % CC;
        out[i] = x[(size_t)(b * TT + TT - 1) * CC + c];
    }
}

// =====================================================================
extern "C" void kernel_launch(void* const* d_in, const int* in_sizes, int n_in,
                              void* d_out, int out_size, void* d_ws, size_t ws_size,
                              hipStream_t stream) {
    const float* x        = (const float*)d_in[0];
    const float* wkv_st   = (const float*)d_in[1];
    const float* shift    = (const float*)d_in[2];
    const float* maa_x    = (const float*)d_in[3];
    const float* maa_r    = (const float*)d_in[4];
    const float* maa_k    = (const float*)d_in[5];
    const float* maa_v    = (const float*)d_in[6];
    const float* maa_w    = (const float*)d_in[7];
    const float* maa_v2   = (const float*)d_in[8];
    const float* maa_w1   = (const float*)d_in[9];
    const float* maa_w2   = (const float*)d_in[10];
    const float* tdecay   = (const float*)d_in[11];
    const float* decay_w1 = (const float*)d_in[12];
    const float* decay_w2 = (const float*)d_in[13];
    const float* val2_w1  = (const float*)d_in[14];
    const float* val2_w2  = (const float*)d_in[15];
    const float* Wr       = (const float*)d_in[16];
    const float* Wk       = (const float*)d_in[17];
    const float* Wv       = (const float*)d_in[18];
    const float* Wo       = (const float*)d_in[19];
    const float* ln_g     = (const float*)d_in[20];
    const float* ln_b     = (const float*)d_in[21];

    // ---- workspace carve-out (~545 MB) ----
    char* p = (char*)d_ws;
    auto alloc = [&](size_t b) { char* r = p; p += (b + 255) & ~(size_t)255; return r; };
    const size_t WW = (size_t)CC * CC;

    unsigned short* bWr = (unsigned short*)alloc(WW * 2);
    unsigned short* bWk = (unsigned short*)alloc(WW * 2);
    unsigned short* bWv = (unsigned short*)alloc(WW * 2);
    unsigned short* bWo = (unsigned short*)alloc(WW * 2);
    unsigned short* d1t = (unsigned short*)alloc((size_t)64 * CC * 2);  // [64][2048]
    unsigned short* d2t = (unsigned short*)alloc((size_t)CC * 64 * 2);  // [2048][64]
    unsigned short* v1t = (unsigned short*)alloc((size_t)64 * CC * 2);
    unsigned short* v2t = (unsigned short*)alloc((size_t)CC * 64 * 2);

    unsigned short* axr  = (unsigned short*)alloc((size_t)BT * CC * 2);
    unsigned short* axk  = (unsigned short*)alloc((size_t)BT * CC * 2);
    unsigned short* axv  = (unsigned short*)alloc((size_t)BT * CC * 2);
    unsigned short* axw  = (unsigned short*)alloc((size_t)BT * CC * 2);
    unsigned short* axv2 = (unsigned short*)alloc((size_t)BT * CC * 2);

    float* xxx    = (float*)alloc((size_t)BT * 160 * 4);
    float* rbuf   = (float*)alloc((size_t)BT * CC * 4);
    float* kbuf   = (float*)alloc((size_t)BT * CC * 4);
    float* vbuf   = (float*)alloc((size_t)BT * CC * 4);
    float* v2buf  = (float*)alloc((size_t)BT * CC * 4);
    float* wdec   = (float*)alloc((size_t)BT * CC * 4);
    unsigned short* lora_a = (unsigned short*)alloc((size_t)BT * 64 * 2);
    unsigned short* lora_b = (unsigned short*)alloc((size_t)BT * 64 * 2);

    // aliases over dead bf16 activations (xr/xk dead after GEMMs; xv dead too)
    float*          ybuf = (float*)axr;                                   // 64 MB
    unsigned short* ylnb = (unsigned short*)((char*)axr + (size_t)BT * CC * 4);

    float* out_y     = (float*)d_out;
    float* out_state = out_y + (size_t)BT * CC;
    float* out_xlast = out_state + (size_t)BB * NHEADS * HH * HH;

    // ---- 1) weights -> bf16 ----
    cvt_bf16_kernel<<<(int)((WW + 255) / 256), 256, 0, stream>>>(Wr, bWr, WW);
    cvt_bf16_kernel<<<(int)((WW + 255) / 256), 256, 0, stream>>>(Wk, bWk, WW);
    cvt_bf16_kernel<<<(int)((WW + 255) / 256), 256, 0, stream>>>(Wv, bWv, WW);
    cvt_bf16_kernel<<<(int)((WW + 255) / 256), 256, 0, stream>>>(Wo, bWo, WW);
    int smallGrid = (CC * 64 + 255) / 256;
    cvt_bf16_t_kernel<<<smallGrid, 256, 0, stream>>>(decay_w1, d1t, CC, 64);
    cvt_bf16_t_kernel<<<smallGrid, 256, 0, stream>>>(decay_w2, d2t, 64, CC);
    cvt_bf16_t_kernel<<<smallGrid, 256, 0, stream>>>(val2_w1,  v1t, CC, 64);
    cvt_bf16_t_kernel<<<smallGrid, 256, 0, stream>>>(val2_w2,  v2t, 64, CC);

    // ---- 2) token shift + LoRA mix ----
    lora_in_kernel<<<BT, 256, 0, stream>>>(x, shift, maa_x, maa_w1, xxx);
    mix_out_kernel<<<BT, 256, 0, stream>>>(x, shift, xxx, maa_w2,
                                           maa_r, maa_k, maa_v, maa_w, maa_v2,
                                           axr, axk, axv, axw, axv2);

    // ---- 3) projections (WMMA bf16, f32 accumulate, async-LDS staging) ----
    dim3 blk(256);
    dim3 gBig(CC / TILE_N, BT / TILE_M);   // (16, 64)
    dim3 gSm(1, BT / TILE_M);              // N=64 fits in one N-tile
    gemm_bf16_wmma<<<gBig, blk, 0, stream>>>(axr,  bWr, CC, CC, rbuf,  nullptr, 0);
    gemm_bf16_wmma<<<gBig, blk, 0, stream>>>(axk,  bWk, CC, CC, kbuf,  nullptr, 0);
    gemm_bf16_wmma<<<gBig, blk, 0, stream>>>(axv,  bWv, CC, CC, vbuf,  nullptr, 0);
    gemm_bf16_wmma<<<gBig, blk, 0, stream>>>(axv2, bWv, CC, CC, v2buf, nullptr, 0);
    // w-branch LoRA: tanh(xw @ decay_w1) @ decay_w2
    gemm_bf16_wmma<<<gSm,  blk, 0, stream>>>(axw, d1t, 64, CC, nullptr, lora_a, 1);
    gemm_bf16_wmma<<<gBig, blk, 0, stream>>>(lora_a, d2t, CC, 64, wdec, nullptr, 0);
    // v2-branch LoRA: v2 += tanh(xv2 @ val2_w1) @ val2_w2
    gemm_bf16_wmma<<<gSm,  blk, 0, stream>>>(axv2, v1t, 64, CC, nullptr, lora_b, 1);
    gemm_bf16_wmma<<<gBig, blk, 0, stream>>>(lora_b, v2t, CC, 64, v2buf, nullptr, 2);

    // ---- 4) decay + k scaling ----
    postmix_kernel<<<(int)(((size_t)BT * CC) / 256), 256, 0, stream>>>(tdecay, wdec, kbuf);

    // ---- 5) serial WKV scan (also emits final state) ----
    wkv_kernel<<<BB * NHEADS, HH, 0, stream>>>(rbuf, kbuf, vbuf, v2buf, wdec,
                                               wkv_st, ybuf, out_state);

    // ---- 6) LayerNorm + output projection ----
    ln_kernel<<<BT, 256, 0, stream>>>(ybuf, ln_g, ln_b, ylnb);
    gemm_bf16_wmma<<<gBig, blk, 0, stream>>>(ylnb, bWo, CC, CC, out_y, nullptr, 0);

    // ---- 7) x[:, -1] ----
    tail_kernel<<<(BB * CC + 255) / 256, 256, 0, stream>>>(x, out_xlast);
}
